// HistoryGRU_58574763982863
// MI455X (gfx1250) — matrix operand order repaired and checked by reading, not compiled, not run on hardware
//
#include <hip/hip_runtime.h>
#include <cstdint>

// ---------------------------------------------------------------------------
// HistoryGRU on MI455X (gfx1250), wave32 + WMMA, compile-time shapes.
//
// DCE insight: reference emits only h_new[-1] per scan step and the GRU row
// update is row-independent => output depends ONLY on history[:, 4095, :].
// 140 GFLOP -> ~35 MFLOP, 436 MB -> ~106 KB. Remaining cost: 128-step
// sequential recurrence (1x131 @ 131x393 per step) in one workgroup with
// W_hh resident in registers as pre-packed WMMA B-fragments, and the hidden
// state staged in LDS as bf16 (2x ds_load_b128 per k-tile, no masking VALU).
// ---------------------------------------------------------------------------

typedef __attribute__((ext_vector_type(16))) __bf16 v16bf;
typedef __attribute__((ext_vector_type(8)))  __bf16 v8bf;
typedef __attribute__((ext_vector_type(8)))  float  v8f;
typedef __attribute__((ext_vector_type(4)))  float  v4f;

#define F_IN 208
#define HH   131
#define G3   393      // 3*H
#define DM   255
#define NOUT 17
#define NSEQ 128
#define TB   4096

// workspace leading dims (floats; all multiples of 4 -> 16B-aligned rows)
#define GI_LD 400
#define X_LD  144
#define A_LD  256

// Fast transcendentals on the sequential critical path: v_rcp_f32/v_exp_f32
// instead of the IEEE div expansion (div_scale + 4x fma Newton).
static __device__ __forceinline__ float fast_sigmoid(float x) {
  return __builtin_amdgcn_rcpf(1.0f + __expf(-x));
}
static __device__ __forceinline__ float fast_tanh(float x) {
  return 2.0f * __builtin_amdgcn_rcpf(1.0f + __expf(-2.0f * x)) - 1.0f;
}

// ---- WMMA fragment gathers -------------------------------------------------

// A fragment 16x32 (rows m0.., cols k0..) from row-major float [*, lda].
// ISA layout: m = lane&15, k = (j>>3)*16 + (lane>>4)*8 + (j&7).
// Fast (unchecked) path requires lda%4==0 and a 16B-aligned base: true for all
// call sites (history stride 851968, GI_LD/X_LD/A_LD above).
template <bool KCHK>
static __device__ __forceinline__ v16bf load_a_frag(const float* __restrict__ A,
                                                    long lda, int m0, int k0, int Klim) {
  const int lane = threadIdx.x & 31;
  const long row = (long)(m0 + (lane & 15)) * lda;
  const int  kb  = (lane >> 4) << 3;
  v16bf a;
  if constexpr (!KCHK) {
    const float* p = A + row + k0 + kb;        // 16B aligned
    const v4f x0 = *(const v4f*)(p);           // k = k0+kb+0..3
    const v4f x1 = *(const v4f*)(p + 4);       // k = k0+kb+4..7
    const v4f x2 = *(const v4f*)(p + 16);      // k = k0+16+kb+0..3
    const v4f x3 = *(const v4f*)(p + 20);      // k = k0+16+kb+4..7
#pragma unroll
    for (int j = 0; j < 4; ++j) {
      a[j]      = (__bf16)x0[j];
      a[j + 4]  = (__bf16)x1[j];
      a[j + 8]  = (__bf16)x2[j];
      a[j + 12] = (__bf16)x3[j];
    }
  } else {
#pragma unroll
    for (int j = 0; j < 16; ++j) {
      const int k = k0 + ((j >> 3) << 4) + kb + (j & 7);
      float v = A[row + min(k, Klim - 1)];
      if (k >= Klim) v = 0.0f;
      a[j] = (__bf16)v;
    }
  }
  return a;
}

// B fragment 32x16 with B[k][n] = W[n0+n][k0+k], W row-major [N, ldw] (A@W^T).
// ISA layout: n = lane&15, k = (j>>3)*16 + (lane>>4)*8 + (j&7). Scalar loads (ldw may be odd).
template <bool NCHK, bool KCHK>
static __device__ __forceinline__ v16bf load_b_fragT(const float* __restrict__ W,
                                                     long ldw, int n0, int k0,
                                                     int Nlim, int Klim) {
  const int lane = threadIdx.x & 31;
  const int n  = n0 + (lane & 15);
  const long row = (long)(NCHK ? min(n, Nlim - 1) : n) * ldw;
  const int  kb  = (lane >> 4) << 3;
  v16bf b;
#pragma unroll
  for (int j = 0; j < 16; ++j) {
    const int k = k0 + ((j >> 3) << 4) + kb + (j & 7);
    float v;
    if (KCHK) {
      v = W[row + min(k, Klim - 1)];
      if (k >= Klim) v = 0.0f;
    } else {
      v = W[row + k];
    }
    if (NCHK && n >= Nlim) v = 0.0f;
    b[j] = (__bf16)v;
  }
  return b;
}

// ---- Generic fused GEMM: C = act(A @ W^T + bias), M == 128 always ----------
template <int N, int K, bool NCHK, int RELU>
static __device__ __forceinline__ void gemm_body(const float* __restrict__ A, long lda,
                                                 const float* __restrict__ W, long ldw,
                                                 const float* __restrict__ bias,
                                                 float* __restrict__ C, long ldc,
                                                 int m0, int n0) {
  v8f acc = {};
#pragma unroll
  for (int kt = 0; kt < K / 32; ++kt) {
    const v16bf a = load_a_frag<false>(A, lda, m0, kt * 32, K);
    const v16bf b = load_b_fragT<NCHK, false>(W, ldw, n0, kt * 32, N, K);
    acc = __builtin_amdgcn_wmma_f32_16x16x32_bf16(false, a, false, b,
                                                  (short)0, acc, false, false);
  }
  if constexpr ((K % 32) != 0) {
    constexpr int k0 = (K / 32) * 32;
    const v16bf a = load_a_frag<true>(A, lda, m0, k0, K);
    const v16bf b = load_b_fragT<NCHK, true>(W, ldw, n0, k0, N, K);
    acc = __builtin_amdgcn_wmma_f32_16x16x32_bf16(false, a, false, b,
                                                  (short)0, acc, false, false);
  }
  const int lane = threadIdx.x & 31;
  const int n  = n0 + (lane & 15);
  const int mr = m0 + ((lane >> 4) << 3);
  if (!NCHK || n < N) {
    const float bv = bias[n];
#pragma unroll
    for (int v = 0; v < 8; ++v) {
      float x = acc[v] + bv;
      if (RELU) x = fmaxf(x, 0.0f);
      C[(long)(mr + v) * ldc + n] = x;
    }
  }
}

// grid = (ceil(N/16), 8), block = 32 (one wave / 16x16 output tile)
template <int N, int K, int RELU>
__global__ void wmma_gemm_bias(const float* __restrict__ A, long lda,
                               const float* __restrict__ W, long ldw,
                               const float* __restrict__ bias,
                               float* __restrict__ C, long ldc) {
  const int n0 = blockIdx.x * 16;
  const int m0 = blockIdx.y * 16;
  if (n0 + 16 <= N)   // uniform branch: EXEC untouched (WMMA needs EXEC all-1s)
    gemm_body<N, K, false, RELU>(A, lda, W, ldw, bias, C, ldc, m0, n0);
  else
    gemm_body<N, K, true, RELU>(A, lda, W, ldw, bias, C, ldc, m0, n0);
}

// ---- GRU recurrence for the single live lane -------------------------------
// One workgroup, 25 waves x 32 = 800 threads. Wave w owns gate columns
// [16w, 16w+16). W_hh lives in registers (5 B-fragments / wave).
//
// A-fragment trick: lanes with m==0 read the bf16 hidden state from sh_hb,
// lanes with m!=0 read from an identically-sized all-zero buffer sh_zero via a
// per-lane base pointer selected ONCE. The per-step fragment build is then
// exactly 2x ds_load_b128 per k-tile: zero VALU masking in the hot loop.
__global__ __launch_bounds__(800)
void gru_lastlane(const float* __restrict__ GI,    // [NSEQ, GI_LD]
                  const float* __restrict__ W_hh,  // [G3, HH] row-major
                  const float* __restrict__ b_hh,  // [G3]
                  float* __restrict__ X) {         // [NSEQ, X_LD] out
  __shared__ __align__(16) __bf16 sh_hb[160];    // bf16 h, zero padded
  __shared__ __align__(16) __bf16 sh_zero[160];  // permanent zeros
  __shared__ float sh_gh[GI_LD];                 // W_hh-path pre-activations

  const int tid  = threadIdx.x;
  const int lane = tid & 31;
  const int n0   = (tid >> 5) * 16;   // wave id * 16 = gate column base

  // One-time preload of this wave's W_hh fragments (K = 131 -> 5 k-tiles).
  v16bf bw[5];
#pragma unroll
  for (int kt = 0; kt < 5; ++kt)
    bw[kt] = load_b_fragT<true, true>(W_hh, HH, n0, kt * 32, G3, HH);

  if (tid < 160) {                    // h0 = 0 (incl. padding) + zero buffer
    sh_hb[tid]   = (__bf16)0.0f;
    sh_zero[tid] = (__bf16)0.0f;
  }
  float hreg = 0.0f;                  // this thread's h[tid]

  // Hoist loop-invariant biases (clamped index keeps all 800 threads in-bounds).
  const int hc = min(tid, HH - 1);
  const float bhr = b_hh[hc], bhz = b_hh[HH + hc], bhn = b_hh[2 * HH + hc];

  // Per-lane A-fragment source: h row for m==0, zeros otherwise.
  const int m  = lane & 15;
  const int kb = (lane >> 4) << 3;                       // 0 or 8
  const __bf16* abase = ((m == 0) ? sh_hb : sh_zero) + kb;

  __syncthreads();

  for (int t = 0; t < NSEQ; ++t) {
    // gh[n0..n0+15] = h @ W_hh^T  (A row 0 = h, rows 1..15 = 0)
    v8f acc = {};
#pragma unroll
    for (int kt = 0; kt < 5; ++kt) {
      const v8bf lo = *(const v8bf*)(abase + kt * 32);        // k = kt*32+kb+0..7
      const v8bf hi = *(const v8bf*)(abase + kt * 32 + 16);   // k = kt*32+16+kb+0..7
      const v16bf a = __builtin_shufflevector(lo, hi, 0, 1, 2, 3, 4, 5, 6, 7,
                                              8, 9, 10, 11, 12, 13, 14, 15);
      acc = __builtin_amdgcn_wmma_f32_16x16x32_bf16(false, a, false, bw[kt],
                                                    (short)0, acc, false, false);
    }
    // Row m==0 -> acc element 0 on lanes 0..15 (n = n0 + lane).
    const int g = n0 + lane;
    if (lane < 16 && g < G3) sh_gh[g] = acc[0];
    __syncthreads();

    // Gate math (f32, fast trans ops). Reads unconditional; writes guarded.
    const float* gi = GI + (long)t * GI_LD;
    const float ir = gi[hc], iz = gi[HH + hc], inn = gi[2 * HH + hc];
    const float hr = sh_gh[hc] + bhr;
    const float hz = sh_gh[HH + hc] + bhz;
    const float hn = sh_gh[2 * HH + hc] + bhn;
    const float r  = fast_sigmoid(ir + hr);
    const float z  = fast_sigmoid(iz + hz);
    const float nn = fast_tanh(inn + r * hn);
    const float hq = (1.0f - z) * nn + z * hreg;
    if (tid < HH) {
      hreg = hq;
      sh_hb[tid] = (__bf16)hq;
      X[(long)t * X_LD + tid] = hq;   // emitted hidden state of the live lane
    }
    __syncthreads();
  }
}

// ---------------------------------------------------------------------------
extern "C" void kernel_launch(void* const* d_in, const int* in_sizes, int n_in,
                              void* d_out, int out_size, void* d_ws, size_t ws_size,
                              hipStream_t stream) {
  const float* history = (const float*)d_in[0];   // [128, 4096, 208]
  const float* W_ih    = (const float*)d_in[1];   // [393, 208]
  const float* W_hh    = (const float*)d_in[2];   // [393, 131]
  const float* b_ih    = (const float*)d_in[3];
  const float* b_hh    = (const float*)d_in[4];
  const float* W1 = (const float*)d_in[5];  const float* b1 = (const float*)d_in[6];
  const float* W2 = (const float*)d_in[7];  const float* b2 = (const float*)d_in[8];
  const float* W3 = (const float*)d_in[9];  const float* b3 = (const float*)d_in[10];
  const float* W4 = (const float*)d_in[11]; const float* b4 = (const float*)d_in[12];
  const float* W5 = (const float*)d_in[13]; const float* b5 = (const float*)d_in[14];
  const float* W6 = (const float*)d_in[15]; const float* b6 = (const float*)d_in[16];
  float* out = (float*)d_out;                     // [128, 17]

  float* GI = (float*)d_ws;                       // [128, 400]
  float* X  = GI + (long)NSEQ * GI_LD;            // [128, 144]
  float* A0 = X  + (long)NSEQ * X_LD;             // [128, 256]
  float* A1 = A0 + (long)NSEQ * A_LD;             // [128, 256]

  // Phase 1: GI = history[:, 4095, :] @ W_ih^T + b_ih   (M=128, N=393, K=208)
  wmma_gemm_bias<G3, F_IN, 0><<<dim3(25, 8), 32, 0, stream>>>(
      history + (long)(TB - 1) * F_IN, (long)TB * F_IN, W_ih, F_IN, b_ih, GI, GI_LD);

  // Phase 2: 128-step GRU recurrence on the single live lane.
  gru_lastlane<<<1, 800, 0, stream>>>(GI, W_hh, b_hh, X);

  // Phase 3: 6-layer MLP on X [128, 131].
  wmma_gemm_bias<DM, HH, 1><<<dim3(16, 8), 32, 0, stream>>>(X,  X_LD, W1, HH, b1, A0, A_LD);
  wmma_gemm_bias<DM, DM, 1><<<dim3(16, 8), 32, 0, stream>>>(A0, A_LD, W2, DM, b2, A1, A_LD);
  wmma_gemm_bias<DM, DM, 1><<<dim3(16, 8), 32, 0, stream>>>(A1, A_LD, W3, DM, b3, A0, A_LD);
  wmma_gemm_bias<DM, DM, 1><<<dim3(16, 8), 32, 0, stream>>>(A0, A_LD, W4, DM, b4, A1, A_LD);
  wmma_gemm_bias<DM, DM, 1><<<dim3(16, 8), 32, 0, stream>>>(A1, A_LD, W5, DM, b5, A0, A_LD);
  wmma_gemm_bias<NOUT, DM, 0><<<dim3(2, 8), 32, 0, stream>>>(A0, A_LD, W6, DM, b6, out, NOUT);
}
